// SSCM_78546361909687
// MI455X (gfx1250) — matrix operand order repaired and checked by reading, not compiled, not run on hardware
//
#include <hip/hip_runtime.h>
#include <hip/hip_bf16.h>
#include <hip/hip_fp16.h>

typedef __attribute__((ext_vector_type(16))) _Float16 v16h;
typedef __attribute__((ext_vector_type(8)))  _Float16 v8h;
typedef __attribute__((ext_vector_type(8)))  float    v8f;

#define BN   4
#define CN   256
#define HN   64
#define WN   64
#define HWN  4096
#define C8N  32
#define C2N  128

// ---------------- WMMA tile helpers (f16 A/B, f32 C/D) ----------------
// A 16x32 f16: lane l -> row M=l&15; halves i<8 : K = kb+i, i>=8 : K = kb+16+(i-8), kb = 8*(l>>4)
__device__ __forceinline__ v16h wmma_load_a(const _Float16* rowp) {
  int kb = ((threadIdx.x & 31) >> 4) * 8;
  v8h lo = *(const v8h*)(rowp + kb);
  v8h hi = *(const v8h*)(rowp + kb + 16);
  v16h a;
#pragma unroll
  for (int i = 0; i < 8; ++i) { a[i] = lo[i]; a[i + 8] = hi[i]; }
  return a;
}
// B 32x16 f16: lane l -> col N=l&15; halves i<8 : K = kb+i, i>=8 : K = kb+16+(i-8), kb = 8*(l>>4)
__device__ __forceinline__ v16h wmma_load_b(const _Float16* colbase, int stride) {
  int l = threadIdx.x & 31;
  const _Float16* p = colbase + (size_t)(l & 15) * (size_t)stride + ((l >> 4) << 3);
  v8h lo = *(const v8h*)p;
  v8h hi = *(const v8h*)(p + 16);
  v16h b;
#pragma unroll
  for (int i = 0; i < 8; ++i) { b[i] = lo[i]; b[i + 8] = hi[i]; }
  return b;
}
__device__ __forceinline__ v8f wmma16x16x32(v16h a, v16h b, v8f c) {
  return __builtin_amdgcn_wmma_f32_16x16x32_f16(false, a, false, b, (short)0, c, false, false);
}
__device__ __forceinline__ v16h zero16h() {
  v16h a;
#pragma unroll
  for (int i = 0; i < 16; ++i) a[i] = (_Float16)0.f;
  return a;
}
#define V8F_ZERO {0.f,0.f,0.f,0.f,0.f,0.f,0.f,0.f}

// ---------------- utility kernels ----------------
// x[B][C][HW] fp32 -> x16 (same layout) + xT16 [B][HW][C], both f16
__global__ void k_cvt_x(const float* __restrict__ x, _Float16* __restrict__ x16,
                        _Float16* __restrict__ xT16) {
  int i = blockIdx.x * 256 + threadIdx.x;      // < B*C*HW = 2^22
  int hw = i & (HWN - 1);
  int c  = (i >> 12) & (CN - 1);
  int b  = i >> 20;
  float v = x[i];
  x16[i] = (_Float16)v;
  xT16[((size_t)b * HWN + hw) * CN + c] = (_Float16)v;
}

// weights: v_w, s_vw straight f16 convert; q_w/k_w [o][c][3][3] -> [tap][o][c] f16
__global__ void k_cvt_w(const float* __restrict__ v_w, const float* __restrict__ s_vw,
                        const float* __restrict__ q_w, const float* __restrict__ k_w,
                        _Float16* vw16, _Float16* svw16, _Float16* qwt16, _Float16* kwt16) {
  int i = blockIdx.x * 256 + threadIdx.x;
  if (i < 65536) {
    vw16[i] = (_Float16)v_w[i];
  } else if (i < 131072) {
    svw16[i - 65536] = (_Float16)s_vw[i - 65536];
  } else if (i < 131072 + 73728) {
    int j = i - 131072; int t = j >> 13; int oc = j & 8191;
    qwt16[j] = (_Float16)q_w[oc * 9 + t];
  } else if (i < 131072 + 147456) {
    int j = i - (131072 + 73728); int t = j >> 13; int oc = j & 8191;
    kwt16[j] = (_Float16)k_w[oc * 9 + t];
  }
}

// ---------------- 3x3 conv (C->32) + bias + BN + ReLU via 9 shifted 1x1 WMMA GEMMs ----------
// out: [B][HW][32] f16.  grid = B*256 blocks, 64 threads (2 waves, one o-tile each)
__global__ void k_conv_qk(const _Float16* __restrict__ xT16, const _Float16* __restrict__ wt16,
                          const float* __restrict__ cb, const float* __restrict__ bns,
                          const float* __restrict__ bnb, const float* __restrict__ bnm,
                          const float* __restrict__ bnv, _Float16* __restrict__ outT) {
  int b  = blockIdx.x >> 8;
  int n0 = (blockIdx.x & 255) << 4;
  int wv = threadIdx.x >> 5;           // 0..1 : o-tile
  int l  = threadIdx.x & 31;
  int n  = n0 + (l & 15);
  int h  = n >> 6, w = n & 63;
  v8f acc = V8F_ZERO;
  for (int t = 0; t < 9; ++t) {
    int dy = t / 3 - 1, dx = t % 3 - 1;
    int hh = h + dy, ww = w + dx;
    bool valid = (hh >= 0) && (hh < HN) && (ww >= 0) && (ww < WN);
    int src = valid ? (hh * WN + ww) : 0;
    const _Float16* rowp = xT16 + ((size_t)b * HWN + src) * CN;
    const _Float16* colb = wt16 + (size_t)(t * C8N + wv * 16) * CN;
    for (int kc = 0; kc < 8; ++kc) {
      v16h a = zero16h();
      if (valid) a = wmma_load_a(rowp + kc * 32);
      v16h bt = wmma_load_b(colb + kc * 32, CN);
      acc = wmma16x16x32(a, bt, acc);
    }
  }
  int col = l & 15, g = l >> 4;
  int o = wv * 16 + col;
  float inv = rsqrtf(bnv[o] + 1e-5f);
  float sc  = bns[o] * inv;
  float sh  = bnb[o] + (cb[o] - bnm[o]) * sc;
#pragma unroll
  for (int r = 0; r < 8; ++r) {
    int row = r + g * 8;
    float y = acc[r] * sc + sh;
    y = y > 0.f ? y : 0.f;
    outT[((size_t)b * HWN + n0 + row) * C8N + o] = (_Float16)y;
  }
}

// ---------------- 1x1 GEMM: v16[B][C][HW] f16 = xT * vw^T + bias ----------------
// grid = B*256 blocks, 256 threads (8 waves, 32 channels each)
__global__ void k_gemm_v(const _Float16* __restrict__ xT16, const _Float16* __restrict__ w16,
                         const float* __restrict__ bias, _Float16* __restrict__ out16) {
  int b  = blockIdx.x >> 8;
  int n0 = (blockIdx.x & 255) << 4;
  int wv = threadIdx.x >> 5;
  int l  = threadIdx.x & 31;
  int c0 = wv * 32;
  const _Float16* rowp = xT16 + ((size_t)b * HWN + n0 + (l & 15)) * CN;
  v8f a0 = V8F_ZERO, a1 = V8F_ZERO;
  for (int kc = 0; kc < 8; ++kc) {
    v16h a  = wmma_load_a(rowp + kc * 32);
    v16h b0 = wmma_load_b(w16 + (size_t)c0 * CN + kc * 32, CN);
    v16h b1 = wmma_load_b(w16 + (size_t)(c0 + 16) * CN + kc * 32, CN);
    a0 = wmma16x16x32(a, b0, a0);
    a1 = wmma16x16x32(a, b1, a1);
  }
  int col = l & 15, g = l >> 4;
#pragma unroll
  for (int t = 0; t < 2; ++t) {
    int o = c0 + t * 16 + col;
    float bo = bias[o];
    v8f acc = t ? a1 : a0;
    v8h st;
#pragma unroll
    for (int r = 0; r < 8; ++r) st[r] = (_Float16)(acc[r] + bo);
    *(v8h*)(out16 + ((size_t)b * CN + o) * HWN + n0 + g * 8) = st;
  }
}

// ---------------- v2 = x*s_vw + s_vb, reduced over channels -> sm[B][2][HW] (max, mean) -----
__global__ void k_v2_sm(const _Float16* __restrict__ xT16, const _Float16* __restrict__ w16,
                        const float* __restrict__ bias, float* __restrict__ sm) {
  __shared__ float lmax[16][8];
  __shared__ float lsum[16][8];
  int b  = blockIdx.x >> 8;
  int n0 = (blockIdx.x & 255) << 4;
  int wv = threadIdx.x >> 5;
  int l  = threadIdx.x & 31;
  int c0 = wv * 32;
  const _Float16* rowp = xT16 + ((size_t)b * HWN + n0 + (l & 15)) * CN;
  v8f a0 = V8F_ZERO, a1 = V8F_ZERO;
  for (int kc = 0; kc < 8; ++kc) {
    v16h a  = wmma_load_a(rowp + kc * 32);
    v16h b0 = wmma_load_b(w16 + (size_t)c0 * CN + kc * 32, CN);
    v16h b1 = wmma_load_b(w16 + (size_t)(c0 + 16) * CN + kc * 32, CN);
    a0 = wmma16x16x32(a, b0, a0);
    a1 = wmma16x16x32(a, b1, a1);
  }
  int col = l & 15, g = l >> 4;
  float bo0 = bias[c0 + col], bo1 = bias[c0 + 16 + col];
#pragma unroll
  for (int r = 0; r < 8; ++r) {
    float v0 = a0[r] + bo0, v1 = a1[r] + bo1;
    float m = fmaxf(v0, v1), s = v0 + v1;
#pragma unroll
    for (int d = 1; d < 16; d <<= 1) {
      m = fmaxf(m, __shfl_xor(m, d, 32));
      s += __shfl_xor(s, d, 32);
    }
    if ((l & 15) == 0) { lmax[r + g * 8][wv] = m; lsum[r + g * 8][wv] = s; }
  }
  __syncthreads();
  if (threadIdx.x < 16) {
    float m = lmax[threadIdx.x][0], s = lsum[threadIdx.x][0];
    for (int j = 1; j < 8; ++j) { m = fmaxf(m, lmax[threadIdx.x][j]); s += lsum[threadIdx.x][j]; }
    sm[(size_t)b * 2 * HWN + n0 + threadIdx.x]       = m;
    sm[(size_t)b * 2 * HWN + HWN + n0 + threadIdx.x] = s * (1.f / CN);
  }
}

// ---------------- fused self-attention: online-softmax flash kernel ----------------
// grid = B*256 blocks (one 16-row n-tile), 256 threads (8 waves).
// Per 256-wide m-chunk: each wave computes its own 16x32 S tile (2 WMMAs, K=32),
// block-reduces row maxima in LDS, rescales f32 accumulators, writes unnormalized
// exp(S - m) to LDS P[16][256], accumulates running denominator, then each wave
// accumulates P x V^T for its 32-channel slice (16 WMMAs). Final divide by l.
__global__ void k_attn_flash(const _Float16* __restrict__ qT, const _Float16* __restrict__ kT,
                             const _Float16* __restrict__ v16p, float* __restrict__ wts) {
  __shared__ __attribute__((aligned(16))) _Float16 P[16][256];
  __shared__ float lmax[16][8];
  __shared__ float lsum[16][8];
  int b  = blockIdx.x >> 8;
  int n0 = (blockIdx.x & 255) << 4;
  int wv = threadIdx.x >> 5, l = threadIdx.x & 31;
  int col = l & 15, g = l >> 4;
  int c0 = wv * 32;
  v16h qa = wmma_load_a(qT + ((size_t)b * HWN + n0 + col) * C8N);
  float mrow[8], lrow[8];
#pragma unroll
  for (int r = 0; r < 8; ++r) { mrow[r] = -1e30f; lrow[r] = 0.f; }
  v8f acc0 = V8F_ZERO, acc1 = V8F_ZERO;
  v8f zz = V8F_ZERO;
  for (int mc = 0; mc < 16; ++mc) {
    int m0 = mc << 8;
    int ms = m0 + wv * 32;
    if (mc + 1 < 16) {   // prefetch next chunk's K and V tiles (global_prefetch_b8)
      __builtin_prefetch(kT + ((size_t)b * HWN + ms + 256) * C8N, 0, 1);
      __builtin_prefetch(v16p + ((size_t)b * CN + c0 + col) * HWN + m0 + 256, 0, 1);
    }
    // --- S tiles for this wave's 32 m-columns ---
    v16h kb0 = wmma_load_b(kT + ((size_t)b * HWN + ms) * C8N, C8N);
    v16h kb1 = wmma_load_b(kT + ((size_t)b * HWN + ms + 16) * C8N, C8N);
    v8f s0 = wmma16x16x32(qa, kb0, zz);
    v8f s1 = wmma16x16x32(qa, kb1, zz);
    // --- per-row tile max -> LDS ---
#pragma unroll
    for (int r = 0; r < 8; ++r) {
      float m = fmaxf(s0[r], s1[r]);
#pragma unroll
      for (int d = 1; d < 16; d <<= 1) m = fmaxf(m, __shfl_xor(m, d, 32));
      if ((l & 15) == 0) lmax[r + g * 8][wv] = m;
    }
    __syncthreads();
    // --- update running max, rescale acc, emit P, partial denominators ---
    float scl[8];
#pragma unroll
    for (int r = 0; r < 8; ++r) {
      int row = r + g * 8;
      float mn = mrow[r];
#pragma unroll
      for (int j = 0; j < 8; ++j) mn = fmaxf(mn, lmax[row][j]);
      float sc = __expf(mrow[r] - mn);
      mrow[r] = mn; scl[r] = sc;
      float p0 = __expf(s0[r] - mn), p1 = __expf(s1[r] - mn);
      P[row][wv * 32 + col]      = (_Float16)p0;
      P[row][wv * 32 + 16 + col] = (_Float16)p1;
      float ps = p0 + p1;
#pragma unroll
      for (int d = 1; d < 16; d <<= 1) ps += __shfl_xor(ps, d, 32);
      if ((l & 15) == 0) lsum[row][wv] = ps;
      acc0[r] *= sc; acc1[r] *= sc;
    }
    __syncthreads();
#pragma unroll
    for (int r = 0; r < 8; ++r) {
      int row = r + g * 8;
      float s = 0.f;
#pragma unroll
      for (int j = 0; j < 8; ++j) s += lsum[row][j];
      lrow[r] = lrow[r] * scl[r] + s;
    }
    // --- P x V^T for this wave's 32 channels ---
#pragma unroll
    for (int ks = 0; ks < 8; ++ks) {
      v16h pa = wmma_load_a(&P[col][ks * 32]);
      v16h b0 = wmma_load_b(v16p + ((size_t)b * CN + c0) * HWN + m0 + ks * 32, HWN);
      v16h b1 = wmma_load_b(v16p + ((size_t)b * CN + c0 + 16) * HWN + m0 + ks * 32, HWN);
      acc0 = wmma16x16x32(pa, b0, acc0);
      acc1 = wmma16x16x32(pa, b1, acc1);
    }
    __syncthreads();
  }
#pragma unroll
  for (int t = 0; t < 2; ++t) {
    int c = c0 + t * 16 + col;
    v8f acc = t ? acc1 : acc0;
    float* dst = wts + ((size_t)b * CN + c) * HWN + n0 + g * 8;
#pragma unroll
    for (int r = 0; r < 8; ++r) dst[r] = acc[r] / lrow[r];
  }
}

// ---------------- channel attention: stats + two-layer FC gate ----------------
__global__ void k_chan_stats(const float* __restrict__ x, float* __restrict__ smean,
                             float* __restrict__ smax) {
  __shared__ float ssum[256], smx[256];
  int bc = blockIdx.x;
  const float* p = x + (size_t)bc * HWN;
  float s = 0.f, m = -1e30f;
  for (int i = threadIdx.x; i < HWN; i += 256) { float v = p[i]; s += v; m = fmaxf(m, v); }
  ssum[threadIdx.x] = s; smx[threadIdx.x] = m;
  __syncthreads();
  for (int d = 128; d > 0; d >>= 1) {
    if (threadIdx.x < d) {
      ssum[threadIdx.x] += ssum[threadIdx.x + d];
      smx[threadIdx.x] = fmaxf(smx[threadIdx.x], smx[threadIdx.x + d]);
    }
    __syncthreads();
  }
  if (threadIdx.x == 0) { smean[bc] = ssum[0] * (1.f / HWN); smax[bc] = smx[0]; }
}

__global__ void k_chan_fc(const float* __restrict__ smean, const float* __restrict__ smax,
                          const float* __restrict__ w1, const float* __restrict__ b1,
                          const float* __restrict__ w2, const float* __restrict__ b2,
                          float* __restrict__ gate) {
  __shared__ float sM[CN], sX[CN], hs[C2N];
  int b = blockIdx.x;
  int t = threadIdx.x;
  sM[t] = smean[b * CN + t];
  sX[t] = smax[b * CN + t];
  __syncthreads();
  if (t < C2N) {
    float am = b1[t], ax = b1[t];
    const float* wr = w1 + (size_t)t * CN;
    for (int c = 0; c < CN; ++c) { am += wr[c] * sM[c]; ax += wr[c] * sX[c]; }
    hs[t] = fmaxf(am, 0.f) + fmaxf(ax, 0.f);
  }
  __syncthreads();
  float gv = 2.f * b2[t];
  const float* wr2 = w2 + (size_t)t * C2N;
  for (int j = 0; j < C2N; ++j) gv += wr2[j] * hs[j];
  gate[b * CN + t] = 1.f / (1.f + __expf(-gv));
}

// ---------------- spatial branch: 2->1 3x3 conv + BN, then row softmax -> aT f16 -----------
__global__ void k_sconv(const float* __restrict__ sm, const float* __restrict__ cw,
                        const float* __restrict__ cb, const float* __restrict__ bs,
                        const float* __restrict__ bb, const float* __restrict__ bm,
                        const float* __restrict__ bv, float* __restrict__ slog) {
  int i = blockIdx.x * 256 + threadIdx.x;         // < B*HW
  int b = i >> 12; int hw = i & 4095; int h = hw >> 6; int w = hw & 63;
  float acc = cb[0];
  for (int ch = 0; ch < 2; ++ch)
    for (int ky = 0; ky < 3; ++ky)
      for (int kx = 0; kx < 3; ++kx) {
        int hh = h + ky - 1, ww = w + kx - 1;
        if (hh >= 0 && hh < HN && ww >= 0 && ww < WN)
          acc += sm[((size_t)b * 2 + ch) * HWN + hh * WN + ww] * cw[ch * 9 + ky * 3 + kx];
      }
  float inv = rsqrtf(bv[0] + 1e-5f);
  slog[i] = (acc - bm[0]) * (inv * bs[0]) + bb[0];
}

__global__ void k_ssoftmax(const float* __restrict__ slog, _Float16* __restrict__ aT) {
  __shared__ float red[64];
  int b = blockIdx.x >> 6, hp = blockIdx.x & 63;
  int k = threadIdx.x;
  float v = slog[(b << 12) + (hp << 6) + k];
  red[k] = v;
  __syncthreads();
  for (int d = 32; d > 0; d >>= 1) {
    if (k < d) red[k] = fmaxf(red[k], red[k + d]);
    __syncthreads();
  }
  float mx = red[0];
  __syncthreads();
  float e = __expf(v - mx);
  red[k] = e;
  __syncthreads();
  for (int d = 32; d > 0; d >>= 1) {
    if (k < d) red[k] += red[k + d];
    __syncthreads();
  }
  float s = red[0];
  aT[(b << 12) + (k << 6) + hp] = (_Float16)(e / s);   // aT[b][k][w']
}

// ---------------- spatialA[b,c,h,k] = sum_w x[b,c,h,w]*a[b,w,k] via WMMA ----------------
// grid = B*C*4 blocks, 64 threads (2 waves, 2 n-tiles each)
__global__ void k_spatial(const _Float16* __restrict__ x16, const _Float16* __restrict__ aT,
                          float* __restrict__ spA) {
  int ht = blockIdx.x & 3;
  int c  = (blockIdx.x >> 2) & 255;
  int b  = blockIdx.x >> 10;
  int wv = threadIdx.x >> 5, l = threadIdx.x & 31;
  int h0 = ht * 16;
  const _Float16* rowp = x16 + ((size_t)b * CN + c) * HWN + (size_t)(h0 + (l & 15)) * WN;
  v8f acc0 = V8F_ZERO, acc1 = V8F_ZERO;
  for (int ks = 0; ks < 2; ++ks) {
    v16h a  = wmma_load_a(rowp + ks * 32);
    v16h b0 = wmma_load_b(aT + (b << 12) + (wv * 2) * 16 * WN + ks * 32, WN);
    v16h b1 = wmma_load_b(aT + (b << 12) + (wv * 2 + 1) * 16 * WN + ks * 32, WN);
    acc0 = wmma16x16x32(a, b0, acc0);
    acc1 = wmma16x16x32(a, b1, acc1);
  }
  int col = l & 15, g = l >> 4;
#pragma unroll
  for (int t = 0; t < 2; ++t) {
    int kk = (wv * 2 + t) * 16 + col;
    v8f acc = t ? acc1 : acc0;
#pragma unroll
    for (int r = 0; r < 8; ++r)
      spA[((size_t)b * CN + c) * HWN + (size_t)(h0 + r + g * 8) * WN + kk] = acc[r];
  }
}

// ---------------- final combine: gamma*wts + x*(1+gate) + spatialA ----------------
__global__ void k_combine(const float* __restrict__ x, const float* __restrict__ wts,
                          const float* __restrict__ spA, const float* __restrict__ gate,
                          const float* __restrict__ gamma, float* __restrict__ out) {
  int i = blockIdx.x * 256 + threadIdx.x;   // < B*C*HW
  int bc = i >> 12;
  out[i] = gamma[0] * wts[i] + x[i] * (1.f + gate[bc]) + spA[i];
}

// ---------------- launch ----------------
extern "C" void kernel_launch(void* const* d_in, const int* in_sizes, int n_in,
                              void* d_out, int out_size, void* d_ws, size_t ws_size,
                              hipStream_t stream) {
  const float* x      = (const float*)d_in[0];
  const float* q_w    = (const float*)d_in[1];
  const float* q_b    = (const float*)d_in[2];
  const float* q_bn_s = (const float*)d_in[3];
  const float* q_bn_b = (const float*)d_in[4];
  const float* q_bn_m = (const float*)d_in[5];
  const float* q_bn_v = (const float*)d_in[6];
  const float* k_w    = (const float*)d_in[7];
  const float* k_b    = (const float*)d_in[8];
  const float* k_bn_s = (const float*)d_in[9];
  const float* k_bn_b = (const float*)d_in[10];
  const float* k_bn_m = (const float*)d_in[11];
  const float* k_bn_v = (const float*)d_in[12];
  const float* v_w    = (const float*)d_in[13];
  const float* v_b    = (const float*)d_in[14];
  const float* gamma  = (const float*)d_in[15];
  const float* c_w1   = (const float*)d_in[16];
  const float* c_b1   = (const float*)d_in[17];
  const float* c_w2   = (const float*)d_in[18];
  const float* c_b2   = (const float*)d_in[19];
  const float* s_vw   = (const float*)d_in[20];
  const float* s_vb   = (const float*)d_in[21];
  const float* s_cw   = (const float*)d_in[22];
  const float* s_cb   = (const float*)d_in[23];
  const float* s_bn_s = (const float*)d_in[24];
  const float* s_bn_b = (const float*)d_in[25];
  const float* s_bn_m = (const float*)d_in[26];
  const float* s_bn_v = (const float*)d_in[27];
  float* out = (float*)d_out;

  char* wsb = (char*)d_ws;
  size_t off = 0;
  auto take = [&](size_t bytes) -> char* {
    size_t a = (off + 255) & ~(size_t)255;
    off = a + bytes;
    return wsb + a;
  };
  const size_t NE = (size_t)BN * CN * HWN;              // 4,194,304
  _Float16* x16   = (_Float16*)take(NE * 2);            // [B][C][HW]
  _Float16* xT16  = (_Float16*)take(NE * 2);            // [B][HW][C]
  _Float16* v16b  = (_Float16*)take(NE * 2);            // [B][C][HW]
  _Float16* qT16  = (_Float16*)take((size_t)BN * HWN * C8N * 2);
  _Float16* kT16  = (_Float16*)take((size_t)BN * HWN * C8N * 2);
  _Float16* vw16  = (_Float16*)take(65536 * 2);
  _Float16* svw16 = (_Float16*)take(65536 * 2);
  _Float16* qwt16 = (_Float16*)take(73728 * 2);
  _Float16* kwt16 = (_Float16*)take(73728 * 2);
  float* wts    = (float*)take(NE * 4);
  float* spA    = (float*)take(NE * 4);
  float* smbuf  = (float*)take((size_t)BN * 2 * HWN * 4);
  float* slog   = (float*)take((size_t)BN * HWN * 4);
  _Float16* aT16 = (_Float16*)take((size_t)BN * HWN * 2);
  float* smean = (float*)take((size_t)BN * CN * 4);
  float* smaxb = (float*)take((size_t)BN * CN * 4);
  float* gateb = (float*)take((size_t)BN * CN * 4);
  (void)in_sizes; (void)n_in; (void)out_size; (void)ws_size;

  // precision conversions / weight re-layout
  k_cvt_x<<<(int)(NE / 256), 256, 0, stream>>>(x, x16, xT16);
  k_cvt_w<<<(131072 + 147456 + 255) / 256, 256, 0, stream>>>(v_w, s_vw, q_w, k_w,
                                                             vw16, svw16, qwt16, kwt16);
  // q/k: 3x3 conv as 9 shifted WMMA GEMMs + BN + ReLU
  k_conv_qk<<<BN * 256, 64, 0, stream>>>(xT16, qwt16, q_b, q_bn_s, q_bn_b, q_bn_m, q_bn_v, qT16);
  k_conv_qk<<<BN * 256, 64, 0, stream>>>(xT16, kwt16, k_b, k_bn_s, k_bn_b, k_bn_m, k_bn_v, kT16);
  // v = x*v_w + v_b (f16, [B][C][HW]); v2 reduced to spatial max/mean maps
  k_gemm_v<<<BN * 256, 256, 0, stream>>>(xT16, vw16, v_b, v16b);
  k_v2_sm<<<BN * 256, 256, 0, stream>>>(xT16, svw16, s_vb, smbuf);
  // fused flash attention (online softmax, exact)
  k_attn_flash<<<BN * 256, 256, 0, stream>>>(qT16, kT16, v16b, wts);
  // channel gate
  k_chan_stats<<<BN * CN, 256, 0, stream>>>(x, smean, smaxb);
  k_chan_fc<<<BN, 256, 0, stream>>>(smean, smaxb, c_w1, c_b1, c_w2, c_b2, gateb);
  // spatial branch
  k_sconv<<<(BN * HWN) / 256, 256, 0, stream>>>(smbuf, s_cw, s_cb, s_bn_s, s_bn_b, s_bn_m,
                                                s_bn_v, slog);
  k_ssoftmax<<<BN * HN, 64, 0, stream>>>(slog, aT16);
  k_spatial<<<BN * CN * 4, 64, 0, stream>>>(x16, aT16, spA);
  // combine
  k_combine<<<(int)(NE / 256), 256, 0, stream>>>(x, wts, spA, gateb, gamma, out);
}
